// GIKT_7215545057916
// MI455X (gfx1250) — compile-verified, matrix operand order checked by reading, not run on hardware
//
#include <hip/hip_runtime.h>
#include <hip/hip_bf16.h>
#include <math.h>

// ---------------- problem constants ----------------
#define PB   16
#define PT   200
#define FD   256            // feature dim
#define NPOS (PB * PT)      // 3200 independent positions

typedef __attribute__((ext_vector_type(8)))  __bf16 v8bf;
typedef __attribute__((ext_vector_type(16))) __bf16 v16bf;
typedef __attribute__((ext_vector_type(8)))  float  v8f;

union BF16x16 { v16bf v; v8bf h[2]; };

__device__ __forceinline__ float sigf(float x) { return 1.0f / (1.0f + expf(-x)); }

// ---------------------------------------------------------------------------
// 16xN = relu/tanh( A(16x256,bf16 in LDS) @ W(N x 256, bf16, row-major)^T + b )
// CDNA5 wave32 WMMA f32_16x16x32_bf16.  8 waves split the N tiles; each wave
// iteration processes TWO N-tiles (nt, nt+8) sharing the A fragments, with the
// B fragments double-buffered across the K loop so >=4 global_load_b128 stay
// in flight per wave while WMMAs execute.
// A layout: lane l (l<16) -> row l, K in {kc*32+0..7, kc*32+16..23};
//           lane l (>=16) -> row l-16, K in {kc*32+8..15, kc*32+24..31}.
// B column n mirrors A row striping; D: lane gives N=l&15, VGPR v gives M=v+8*(l>>4).
// act: 0 = none, 1 = relu, 2 = tanh
// ---------------------------------------------------------------------------
__device__ __forceinline__ void gemm_tile(const __bf16* __restrict__ W,
                                          const float* __restrict__ bias,
                                          const __bf16* sA,
                                          float* out, int outStride,
                                          int N, int validM, int act)
{
    const int tid   = threadIdx.x;
    const int wave  = tid >> 5;
    const int lane  = tid & 31;
    const int ncol  = lane & 15;
    const int mbase = (lane >> 4) << 3;      // 0 or 8 (C/D row offset)
    const int khalf = (lane >> 4) << 3;      // 0 or 8 (K sub-block)
    const int ntiles = N >> 4;               // 16 (N=256) or 64 (N=1024)

    const __bf16* arow = sA + (lane & 15) * FD;

    for (int nt0 = wave; nt0 < ntiles; nt0 += 16) {
        const int nt1 = nt0 + 8;
        const int n0  = nt0 * 16 + ncol;
        const int n1  = nt1 * 16 + ncol;
        const __bf16* brow0 = W + (size_t)n0 * FD;
        const __bf16* brow1 = W + (size_t)n1 * FD;

        v8f acc0 = {}, acc1 = {};

        // prologue: B fragments for kc = 0 (both tiles)
        BF16x16 b0c, b1c;
        b0c.h[0] = *(const v8bf*)(brow0 + khalf);
        b0c.h[1] = *(const v8bf*)(brow0 + khalf + 16);
        b1c.h[0] = *(const v8bf*)(brow1 + khalf);
        b1c.h[1] = *(const v8bf*)(brow1 + khalf + 16);

#pragma unroll
        for (int kc = 0; kc < 8; ++kc) {
            // prefetch next K-chunk's B fragments before issuing this WMMA pair
            BF16x16 b0n = b0c, b1n = b1c;
            if (kc < 7) {
                const int ko = (kc + 1) * 32 + khalf;
                b0n.h[0] = *(const v8bf*)(brow0 + ko);
                b0n.h[1] = *(const v8bf*)(brow0 + ko + 16);
                b1n.h[0] = *(const v8bf*)(brow1 + ko);
                b1n.h[1] = *(const v8bf*)(brow1 + ko + 16);
            }
            BF16x16 a;
            a.h[0] = *(const v8bf*)(arow + kc * 32 + khalf);
            a.h[1] = *(const v8bf*)(arow + kc * 32 + khalf + 16);

            acc0 = __builtin_amdgcn_wmma_f32_16x16x32_bf16(
                       false, a.v, false, b0c.v, (short)0, acc0, false, false);
            acc1 = __builtin_amdgcn_wmma_f32_16x16x32_bf16(
                       false, a.v, false, b1c.v, (short)0, acc1, false, false);
            b0c = b0n;
            b1c = b1n;
        }

        const float bn0 = bias ? bias[n0] : 0.0f;
        const float bn1 = bias ? bias[n1] : 0.0f;
#pragma unroll
        for (int v = 0; v < 8; ++v) {
            const int m = v + mbase;
            if (m < validM) {
                float r0 = acc0[v] + bn0;
                float r1 = acc1[v] + bn1;
                if (act == 1)      { r0 = fmaxf(r0, 0.0f); r1 = fmaxf(r1, 0.0f); }
                else if (act == 2) { r0 = tanhf(r0);       r1 = tanhf(r1); }
                out[m * outStride + n0] = r0;
                out[m * outStride + n1] = r1;
            }
        }
    }
}

// ---------------------------------------------------------------------------
// Fused per-position kernel: gathers, 6 aggregation GEMMs, last/ft, LSTM -> h
// One block (8 waves) per (b,t) position.
// ---------------------------------------------------------------------------
__global__ void __launch_bounds__(256)
fused_agg_kernel(const int* __restrict__ question, const int* __restrict__ response,
                 const int* __restrict__ maskv,
                 const int* __restrict__ qnbr, const int* __restrict__ snbr,
                 const float* __restrict__ emb_q, const float* __restrict__ emb_s,
                 const __bf16* __restrict__ wAgg, const float* __restrict__ aggB,
                 const __bf16* __restrict__ wLast, const float* __restrict__ lastB,
                 const __bf16* __restrict__ wFt, const float* __restrict__ ftB,
                 const __bf16* __restrict__ wLstm, const float* __restrict__ rbias,
                 float* __restrict__ h_out)
{
    __shared__ int   sI1[4], sI2[16], sI3[64];
    __shared__ float sL0[FD];           // original emb_q[question]
    __shared__ float sE0[FD];           // embs[0]
    __shared__ float sE1[4 * FD];       // embs[1]
    __shared__ float sE2[16 * FD];      // embs[2]
    __shared__ float sT[16 * FD];       // sum of emb_s[n3] per group / scratch row
    __shared__ float sG[1024];          // LSTM gates
    __shared__ __align__(16) __bf16 sA[16 * FD];

    const int pos = blockIdx.x;
    const int tid = threadIdx.x;
    const int q   = question[pos];

    // Warm L2 with the stage-0 aggregation weights (global_prefetch_b8).
    for (int i = tid; i < 1024; i += 256)
        __builtin_prefetch((const char*)wAgg + (size_t)i * 128, 0, 1);

    // ---- neighbor indices ----
    if (tid < 4) sI1[tid] = qnbr[q * 4 + tid];
    __syncthreads();
    if (tid < 16) sI2[tid] = snbr[sI1[tid >> 2] * 4 + (tid & 3)];
    sL0[tid] = emb_q[(size_t)q * FD + tid];
    for (int i = tid; i < 4 * FD; i += 256)
        sE1[i] = emb_s[(size_t)sI1[i >> 8] * FD + (i & 255)];
    __syncthreads();
    if (tid < 64) sI3[tid] = qnbr[sI2[tid >> 2] * 4 + (tid & 3)];
    for (int i = tid; i < 16 * FD; i += 256)
        sE2[i] = emb_q[(size_t)sI2[i >> 8] * FD + (i & 255)];
    for (int i = tid; i < 16 * FD; i += 256) sT[i] = 0.0f;
    __syncthreads();
    // sum emb_s[n3] rows into 16 group rows (thread tid owns column tid)
    for (int r = 0; r < 64; ++r)
        sT[(r >> 2) * FD + tid] += emb_s[(size_t)sI3[r] * FD + tid];
    __syncthreads();

    // ---- stage (i=0, j=0): x0 = L0 + mean(E1) ; 1 row, W0, relu -> E0 ----
    for (int i = FD + tid; i < 16 * FD; i += 256) sA[i] = (__bf16)0.0f; // clear rows 1..15
    {
        float m = 0.25f * (sE1[tid] + sE1[FD + tid] + sE1[2 * FD + tid] + sE1[3 * FD + tid]);
        sA[tid] = (__bf16)(sL0[tid] + m);
    }
    __syncthreads();
    gemm_tile(wAgg, aggB, sA, sE0, FD, FD, 1, 1);
    __syncthreads();

    // ---- stage (0,1): x1 = E1 + mean4(E2) ; 4 rows, W1, relu -> E1 ----
    for (int i = tid; i < 4 * FD; i += 256) {
        int j = i >> 8, k = i & 255;
        float m = 0.25f * (sE2[(4 * j) * FD + k] + sE2[(4 * j + 1) * FD + k] +
                           sE2[(4 * j + 2) * FD + k] + sE2[(4 * j + 3) * FD + k]);
        sA[i] = (__bf16)(sE1[i] + m);
    }
    __syncthreads();
    gemm_tile(wAgg + 65536, aggB + 256, sA, sE1, FD, FD, 4, 1);
    __syncthreads();

    // ---- stage (0,2): x2 = E2 + 0.25*sumE3 ; 16 rows, W2, relu -> E2 ----
    for (int i = tid; i < 16 * FD; i += 256)
        sA[i] = (__bf16)(sE2[i] + 0.25f * sT[i]);
    __syncthreads();
    gemm_tile(wAgg + 2 * 65536, aggB + 512, sA, sE2, FD, FD, 16, 1);
    __syncthreads();

    // ---- stage (1,0): x0 = E0 + mean(E1) ; 1 row, W0, relu -> E0 ----
    for (int i = FD + tid; i < 16 * FD; i += 256) sA[i] = (__bf16)0.0f;
    {
        float m = 0.25f * (sE1[tid] + sE1[FD + tid] + sE1[2 * FD + tid] + sE1[3 * FD + tid]);
        sA[tid] = (__bf16)(sE0[tid] + m);
    }
    __syncthreads();
    gemm_tile(wAgg, aggB, sA, sE0, FD, FD, 1, 1);
    __syncthreads();

    // ---- stage (1,1): x1 = E1 + mean4(E2) ; 4 rows, W1, relu -> E1 ----
    for (int i = tid; i < 4 * FD; i += 256) {
        int j = i >> 8, k = i & 255;
        float m = 0.25f * (sE2[(4 * j) * FD + k] + sE2[(4 * j + 1) * FD + k] +
                           sE2[(4 * j + 2) * FD + k] + sE2[(4 * j + 3) * FD + k]);
        sA[i] = (__bf16)(sE1[i] + m);
    }
    __syncthreads();
    gemm_tile(wAgg + 65536, aggB + 256, sA, sE1, FD, FD, 4, 1);
    __syncthreads();

    // ---- stage (2,0): x0 = E0 + mean(E1) ; 1 row, W0, relu -> E0 ----
    for (int i = FD + tid; i < 16 * FD; i += 256) sA[i] = (__bf16)0.0f;
    {
        float m = 0.25f * (sE1[tid] + sE1[FD + tid] + sE1[2 * FD + tid] + sE1[3 * FD + tid]);
        sA[tid] = (__bf16)(sE0[tid] + m);
    }
    __syncthreads();
    gemm_tile(wAgg, aggB, sA, sE0, FD, FD, 1, 1);
    __syncthreads();

    // ---- agg = tanh(E0 @ lastW^T + last_b) -> sT row 0 ----
    sA[tid] = (__bf16)sE0[tid];
    __syncthreads();
    gemm_tile(wLast, lastB, sA, sT, FD, FD, 1, 2);
    __syncthreads();

    // ---- emb_question = mask ? agg : L0 ; q_trans = relu(@ftW^T + ft_b) -> E0 ----
    {
        const int mk = maskv[pos];
        float eq = (mk == 1) ? sT[tid] : sL0[tid];
        sA[tid] = (__bf16)eq;
    }
    __syncthreads();
    gemm_tile(wFt, ftB, sA, sE0, FD, FD, 1, 1);
    __syncthreads();

    // ---- LSTM gates = q_trans @ WihL^T + rbias[response] (N=1024) ----
    sA[tid] = (__bf16)sE0[tid];
    __syncthreads();
    {
        const int resp = response[pos];
        gemm_tile(wLstm, rbias + resp * 1024, sA, sG, 0, 1024, 1, 0);
    }
    __syncthreads();

    // ---- h = sigmoid(o) * tanh(sigmoid(i) * tanh(g)) ----
    {
        float ig = sG[tid];
        float gg = sG[512 + tid];
        float og = sG[768 + tid];
        float c  = sigf(ig) * tanhf(gg);
        float h  = sigf(og) * tanhf(c);
        h_out[(size_t)pos * FD + tid] = h;
    }
}

// ---------------------------------------------------------------------------
// Collapsed attention scorer: one wave per output position.
// y[b,t] = sigmoid(disc*(emb_q[q_t]·u_q + h[b,t-1]·u_k + const) + diff_bias[q_t])
// ---------------------------------------------------------------------------
__global__ void __launch_bounds__(256)
score_kernel(const int* __restrict__ question, const float* __restrict__ h,
             const float* __restrict__ emb_q, const float* __restrict__ uvec,
             const float* __restrict__ diff_bias, const float* __restrict__ disc,
             float* __restrict__ y)
{
    const int gidx = blockIdx.x * 8 + (threadIdx.x >> 5);
    const int lane = threadIdx.x & 31;
    if (gidx >= NPOS) return;
    const int t = gidx % PT;
    if (t == 0) { if (lane == 0) y[gidx] = 0.0f; return; }

    const int qn = question[gidx];
    const float* eq = emb_q + (size_t)qn * FD;
    const float* hv = h + (size_t)(gidx - 1) * FD;

    float s = 0.0f;
    for (int k = lane; k < FD; k += 32)
        s += eq[k] * uvec[k] + hv[k] * uvec[FD + k];
#pragma unroll
    for (int off = 16; off > 0; off >>= 1) s += __shfl_xor(s, off, 32);

    if (lane == 0) {
        float logit = s + uvec[512];
        float z = disc[0] * logit + diff_bias[qn];
        y[gidx] = 1.0f / (1.0f + expf(-z));
    }
}

// ---------------- prep kernels (run once per launch) ----------------
__global__ void convert_kernel(const float* __restrict__ src, __bf16* __restrict__ dst, int n)
{
    for (int i = blockIdx.x * blockDim.x + threadIdx.x; i < n; i += gridDim.x * blockDim.x)
        dst[i] = (__bf16)src[i];
}

// left half (columns 0..255) of lstm_Wih (1024 x 512) -> bf16 (1024 x 256)
__global__ void convert_lstm_kernel(const float* __restrict__ Wih, __bf16* __restrict__ dst)
{
    int i = blockIdx.x * blockDim.x + threadIdx.x;
    if (i < 1024 * 256) {
        int n = i >> 8, k = i & 255;
        dst[i] = (__bf16)Wih[n * 512 + k];
    }
}

// rbias[r][n] = Wih[n,256:512]·emb_r[r] + bih[n] + bhh[n]
__global__ void rbias_kernel(const float* __restrict__ Wih, const float* __restrict__ emb_r,
                             const float* __restrict__ bih, const float* __restrict__ bhh,
                             float* __restrict__ rbias)
{
    int i = blockIdx.x * blockDim.x + threadIdx.x;
    if (i < 2048) {
        int r = i >> 10, n = i & 1023;
        const float* w  = Wih + n * 512 + 256;
        const float* er = emb_r + r * FD;
        float s = bih[n] + bhh[n];
        for (int k = 0; k < FD; ++k) s += w[k] * er[k];
        rbias[i] = s;
    }
}

// u_q = query_W^T w[:256], u_k = key_W^T w[256:], const = q_b·w_q + k_b·w_k + W_b
__global__ void uvec_kernel(const float* __restrict__ qW, const float* __restrict__ qb,
                            const float* __restrict__ kW, const float* __restrict__ kb,
                            const float* __restrict__ WW, const float* __restrict__ Wb,
                            float* __restrict__ u)
{
    int k = threadIdx.x;
    float sq = 0.0f, sk = 0.0f;
    for (int n = 0; n < FD; ++n) {
        sq += qW[n * FD + k] * WW[n];
        sk += kW[n * FD + k] * WW[FD + n];
    }
    u[k] = sq;
    u[FD + k] = sk;
    if (k == 0) {
        float c = Wb[0];
        for (int n = 0; n < FD; ++n) c += qb[n] * WW[n] + kb[n] * WW[FD + n];
        u[512] = c;
    }
}

// ---------------------------------------------------------------------------
extern "C" void kernel_launch(void* const* d_in, const int* in_sizes, int n_in,
                              void* d_out, int out_size, void* d_ws, size_t ws_size,
                              hipStream_t stream)
{
    const int*   question  = (const int*)  d_in[0];
    const int*   response  = (const int*)  d_in[1];
    const int*   maskv     = (const int*)  d_in[2];
    const int*   qnbr      = (const int*)  d_in[3];
    const int*   snbr      = (const int*)  d_in[4];
    const float* emb_q     = (const float*)d_in[5];
    const float* emb_s     = (const float*)d_in[6];
    const float* emb_r     = (const float*)d_in[7];
    const float* ft_W      = (const float*)d_in[8];
    const float* ft_b      = (const float*)d_in[9];
    const float* agg_Ws    = (const float*)d_in[10];
    const float* agg_bs    = (const float*)d_in[11];
    const float* last_W    = (const float*)d_in[12];
    const float* last_b    = (const float*)d_in[13];
    const float* lstm_Wih  = (const float*)d_in[14];
    const float* lstm_bih  = (const float*)d_in[15];
    const float* lstm_bhh  = (const float*)d_in[16];
    const float* query_W   = (const float*)d_in[17];
    const float* query_b   = (const float*)d_in[18];
    const float* key_W     = (const float*)d_in[19];
    const float* key_b     = (const float*)d_in[20];
    const float* W_W       = (const float*)d_in[21];
    const float* W_b       = (const float*)d_in[22];
    const float* diff_bias = (const float*)d_in[23];
    const float* disc      = (const float*)d_in[24];

    char* ws = (char*)d_ws;
    __bf16* wAgg  = (__bf16*)(ws);                 // 3*65536 bf16 = 393216 B
    __bf16* wLast = (__bf16*)(ws + 393216);        // 131072 B
    __bf16* wFt   = (__bf16*)(ws + 524288);        // 131072 B
    __bf16* wLstm = (__bf16*)(ws + 655360);        // 524288 B
    float*  rbias = (float*) (ws + 1179648);       // 8192 B
    float*  uvec  = (float*) (ws + 1187840);       // 4096 B
    float*  hbuf  = (float*) (ws + 1191936);       // NPOS*256*4 = 3276800 B
    float*  y     = (float*)d_out;

    convert_kernel<<<768, 256, 0, stream>>>(agg_Ws, wAgg, 3 * 65536);
    convert_kernel<<<256, 256, 0, stream>>>(last_W, wLast, 65536);
    convert_kernel<<<256, 256, 0, stream>>>(ft_W, wFt, 65536);
    convert_lstm_kernel<<<1024, 256, 0, stream>>>(lstm_Wih, wLstm);
    rbias_kernel<<<8, 256, 0, stream>>>(lstm_Wih, emb_r, lstm_bih, lstm_bhh, rbias);
    uvec_kernel<<<1, 256, 0, stream>>>(query_W, query_b, key_W, key_b, W_W, W_b, uvec);

    fused_agg_kernel<<<NPOS, 256, 0, stream>>>(
        question, response, maskv, qnbr, snbr, emb_q, emb_s,
        wAgg, agg_bs, wLast, last_b, wFt, ft_b, wLstm, rbias, hbuf);

    score_kernel<<<(NPOS + 7) / 8, 256, 0, stream>>>(
        question, hbuf, emb_q, uvec, diff_bias, disc, y);
}
